// TaxoRelOLP_48103633715707
// MI455X (gfx1250) — matrix operand order repaired and checked by reading, not compiled, not run on hardware
//
#include <hip/hip_runtime.h>
#include <hip/hip_bf16.h>
#include <math.h>

typedef __attribute__((ext_vector_type(16))) __bf16 v16bf;
typedef __attribute__((ext_vector_type(8)))  float  v8f;

#define BG   4096   // graphs per side
#define NPG  32     // nodes per graph
#define EPG  128    // edges per graph
#define DD   256    // emb dim
#define DD4  64     // hidden dim
// LDS pitches (bf16 elements). Row strides in dwords mod 64 banks:
// 264 -> 132 = 4 mod 64; 72 -> 36 mod 64; 40 -> 20 mod 64.  All give 16
// distinct bank groups for the 16 rows of a fragment => conflict-free b128.
#define HP   264    // node-feature A operand (32 x 256)
#define AP   72     // layer-2 A operands     (32 x 64)
#define TP   40     // transposed B operands  (<=64 x 32) and adjacency (32 x 32)

union FragCast { struct { uint4 a, b; } u; v16bf v; };
union BF2U { struct { __bf16 lo, hi; } h; unsigned u; };
union BF4U { __bf16 h[4]; uint2 u; };

__device__ __forceinline__ v8f wmma_bf16(v16bf a, v16bf b, v8f c) {
  return __builtin_amdgcn_wmma_f32_16x16x32_bf16(
      false, a, false, b, (short)0, c, false, false);
}

// A fragment (16x32 bf16) from LDS, row pitch P: two ds_load_b128, no cvt.
// lane L -> row m_base+(L&15); 16 consecutive K at k_base+kh (kh=0|16).
template <int P>
__device__ __forceinline__ v16bf load_a_frag_lds(const __bf16* src, int m_base,
                                                 int k_base, int lane) {
  const int row = m_base + (lane & 15);
  const int kh  = (lane & 16) ? 16 : 0;
  const __bf16* p = src + row * P + k_base + kh;
  FragCast f;
  f.u.a = *(const uint4*)(p);
  f.u.b = *(const uint4*)(p + 8);
  return f.v;
}

// B fragment (32x16 bf16) from a TRANSPOSED LDS buffer T[n][k] (pitch P):
// lane L -> col n_base+(L&15); 16 consecutive K at k0 = (L&16)?16:0.
template <int P>
__device__ __forceinline__ v16bf load_b_frag_lds(const __bf16* T, int n_base,
                                                 int lane) {
  const int n  = n_base + (lane & 15);
  const int k0 = (lane & 16) ? 16 : 0;
  const __bf16* p = T + n * P + k0;
  FragCast f;
  f.u.a = *(const uint4*)(p);
  f.u.b = *(const uint4*)(p + 8);
  return f.v;
}

// B fragment from pre-swizzled bf16 weights in global: 2x coalesced b128.
__device__ __forceinline__ v16bf load_b_frag_pre(const __bf16* __restrict__ Wf,
                                                 int kst, int nt, int kk, int lane) {
  const uint4* p = (const uint4*)(Wf + (((size_t)(nt * kst + kk) * 32 + lane) << 4));
  FragCast f;
  f.u.a = p[0];
  f.u.b = p[1];
  return f.v;
}

// Store a 16x16 f32 accum tile TRANSPOSED as packed bf16 into T[n][m] (pitch TP).
__device__ __forceinline__ void store_tile_T(const v8f& c, __bf16* T, int n_base,
                                             int m_base, int lane) {
  const int trow = n_base + (lane & 15);
  const int tcol = m_base + ((lane & 16) ? 8 : 0);
  __bf16* p = T + trow * TP + tcol;
#pragma unroll
  for (int j = 0; j < 8; j += 2) {
    BF2U q;
    q.h.lo = (__bf16)c[j];
    q.h.hi = (__bf16)c[j + 1];
    *(unsigned*)(p + j) = q.u;
  }
}

// ---------------------------------------------------------------------------
// Kernel 0: swizzle a f32 weight (N out x K in, row-major) into bf16
// B-fragment order [ntile][kstep][lane][16].
// ---------------------------------------------------------------------------
__global__ __launch_bounds__(256) void prep_weight_kernel(
    const float* __restrict__ W, __bf16* __restrict__ Wf,
    int ntiles, int kst, int K) {
  const int idx = blockIdx.x * 256 + threadIdx.x;
  if (idx >= ntiles * kst * 32) return;
  const int lane = idx & 31;
  const int frag = idx >> 5;
  const int nt = frag / kst;
  const int kk = frag - nt * kst;
  const int col = nt * 16 + (lane & 15);
  const int k0  = kk * 32 + ((lane & 16) ? 16 : 0);
  const float4* p = (const float4*)(W + (size_t)col * K + k0);
  float4 q0 = p[0], q1 = p[1], q2 = p[2], q3 = p[3];
  __bf16* d = Wf + ((size_t)frag * 32 + lane) * 16;
  d[0]=(__bf16)q0.x;  d[1]=(__bf16)q0.y;  d[2]=(__bf16)q0.z;  d[3]=(__bf16)q0.w;
  d[4]=(__bf16)q1.x;  d[5]=(__bf16)q1.y;  d[6]=(__bf16)q1.z;  d[7]=(__bf16)q1.w;
  d[8]=(__bf16)q2.x;  d[9]=(__bf16)q2.y;  d[10]=(__bf16)q2.z; d[11]=(__bf16)q2.w;
  d[12]=(__bf16)q3.x; d[13]=(__bf16)q3.y; d[14]=(__bf16)q3.z; d[15]=(__bf16)q3.w;
}

// ---------------------------------------------------------------------------
// Kernel 1: per-graph CompGCN x2 + mean readout, fully as WMMA:
//   A_f = D_f^-1 C_f, A_r = D_r^-1 C_r  (32x32 normalized adjacency, bf16)
//   h1  = relu(A_f (h @ WO1^T) + A_r (h @ WI1^T) + b1)      [GEMM-first]
//   h2  = relu((A_f h1) @ WO2^T + (A_r h1) @ WI2^T + b2)    [agg-first]
// One block per graph; blockIdx.x in [0, 2*BG): side 0 = subj, 1 = obj.
// ---------------------------------------------------------------------------
__global__ __launch_bounds__(256) void graph_emb_kernel(
    const float* __restrict__ subj_nodes, const float* __restrict__ obj_nodes,
    const int* __restrict__ subj_src, const int* __restrict__ subj_dst,
    const int* __restrict__ obj_src,  const int* __restrict__ obj_dst,
    const __bf16* __restrict__ WO1f, const float* __restrict__ bO1,
    const __bf16* __restrict__ WI1f, const float* __restrict__ bI1,
    const __bf16* __restrict__ WO2f, const float* __restrict__ bO2,
    const __bf16* __restrict__ WI2f, const float* __restrict__ bI2,
    float* __restrict__ out_embs) {
  __shared__ __attribute__((aligned(16))) __bf16 sH[NPG * HP];    // node feats (A op)
  __shared__ __attribute__((aligned(16))) float  sOut[NPG * DD];  // final h2 (f32)
  __shared__ __attribute__((aligned(16))) float  sAF[NPG * NPG];  // fwd counts
  __shared__ __attribute__((aligned(16))) float  sAR[NPG * NPG];  // rev counts
  __shared__ __attribute__((aligned(16))) __bf16 sAFb[NPG * TP];  // A_f bf16 (A op)
  __shared__ __attribute__((aligned(16))) __bf16 sARb[NPG * TP];  // A_r bf16 (A op)
  __shared__ __attribute__((aligned(16))) __bf16 sPOT[DD4 * TP];  // (h@WO1^T)^T (B op)
  __shared__ __attribute__((aligned(16))) __bf16 sPIT[DD4 * TP];  // (h@WI1^T)^T (B op)
  __shared__ __attribute__((aligned(16))) __bf16 sH1T[DD4 * TP];  // h1^T (B op)
  __shared__ __attribute__((aligned(16))) __bf16 sHO1[NPG * AP];  // A_f h1 (A op)
  __shared__ __attribute__((aligned(16))) __bf16 sHI1[NPG * AP];  // A_r h1 (A op)
  __shared__ float sRF[NPG];   // 1/max(deg_fwd,1)
  __shared__ float sRR[NPG];   // 1/max(deg_rev,1)

  const int tid  = threadIdx.x;
  const int lane = tid & 31;
  const int wave = tid >> 5;
  const int bid  = blockIdx.x;
  const int side = bid >> 12;        // / BG
  const int g    = bid & (BG - 1);

  const float* nodes = side ? obj_nodes : subj_nodes;
  const int*   esrc  = side ? obj_src  : subj_src;
  const int*   edst  = side ? obj_dst  : subj_dst;

  if (tid < 4) {
    const __bf16* w = (tid == 0) ? WO1f : (tid == 1) ? WI1f : (tid == 2) ? WO2f : WI2f;
    __builtin_prefetch(w, 0, 0);
  }

  // ---- Stage: node feats f32->bf16 into sH; zero count matrices -----------
  {
    const float4* gsrc = (const float4*)(nodes + (size_t)g * NPG * DD);
#pragma unroll
    for (int i = 0; i < 8; ++i) {
      const int i4  = tid + i * 256;        // float4 index
      const int idx = i4 * 4;
      const int row = idx >> 8;
      const int col = idx & 255;
      const float4 q = gsrc[i4];
      BF4U b;
      b.h[0] = (__bf16)q.x; b.h[1] = (__bf16)q.y;
      b.h[2] = (__bf16)q.z; b.h[3] = (__bf16)q.w;
      *(uint2*)&sH[row * HP + col] = b.u;
    }
    float4* f4 = (float4*)sAF;
    float4* r4 = (float4*)sAR;
    const float4 z = make_float4(0.f, 0.f, 0.f, 0.f);
#pragma unroll
    for (int i = 0; i < 1; ++i) { f4[tid] = z; r4[tid] = z; }  // 256*16B = 4KB each
    if (tid < NPG) { sRF[tid] = 0.f; sRR[tid] = 0.f; }
  }
  __syncthreads();

  // ---- adjacency counts + degrees (512 scalar LDS atomics total) ----------
  if (tid < EPG) {
    const int base = g * NPG;
    const int s = esrc[(size_t)g * EPG + tid] - base;
    const int d = edst[(size_t)g * EPG + tid] - base;
    atomicAdd(&sAF[d * NPG + s], 1.0f);
    atomicAdd(&sAR[s * NPG + d], 1.0f);
    atomicAdd(&sRF[d], 1.0f);
    atomicAdd(&sRR[s], 1.0f);
  }
  __syncthreads();
  if (tid < NPG) {
    sRF[tid] = 1.0f / fmaxf(sRF[tid], 1.0f);
    sRR[tid] = 1.0f / fmaxf(sRR[tid], 1.0f);
  }
  __syncthreads();

  // ---- normalize adjacency -> bf16 A-operand layout (runs with GEMM below)
#pragma unroll
  for (int i = 0; i < 2; ++i) {
    const int idx = i * 512 + tid * 2;   // 1024 elements each
    const int row = idx >> 5;
    const int col = idx & 31;
    const float rf = sRF[row], rr = sRR[row];
    BF2U f, r;
    f.h.lo = (__bf16)(sAF[idx] * rf); f.h.hi = (__bf16)(sAF[idx + 1] * rf);
    r.h.lo = (__bf16)(sAR[idx] * rr); r.h.hi = (__bf16)(sAR[idx + 1] * rr);
    *(unsigned*)&sAFb[row * TP + col] = f.u;
    *(unsigned*)&sARb[row * TP + col] = r.u;
  }

  // ---- Step 2: P_O = h@WO1^T, P_I = h@WI1^T (32x64 each), stored transposed
  {
#pragma unroll
    for (int q = 0; q < 2; ++q) {
      const int job = wave * 2 + q;           // 16 tile jobs
      const int mat = job >> 3;               // 0: WO1, 1: WI1 (uniform per wave)
      const int mt  = (job >> 2) & 1;
      const int nt  = job & 3;
      const __bf16* Wf = mat ? WI1f : WO1f;
      v8f c = {};
#pragma unroll
      for (int kk = 0; kk < 8; ++kk)
        c = wmma_bf16(load_a_frag_lds<HP>(sH, mt * 16, kk * 32, lane),
                      load_b_frag_pre(Wf, 8, nt, kk, lane), c);
      store_tile_T(c, mat ? sPIT : sPOT, nt * 16, mt * 16, lane);
    }
  }
  __syncthreads();

  // ---- Step 4: h1 = relu(A_f P_O + A_r P_I + b1) -> h1^T (bf16) -----------
  {
    const int nt = wave & 3;    // 4 col tiles (64 feats)
    const int mt = wave >> 2;   // 2 row tiles (32 nodes)
    v8f c = {};
    c = wmma_bf16(load_a_frag_lds<TP>(sAFb, mt * 16, 0, lane),
                  load_b_frag_lds<TP>(sPOT, nt * 16, lane), c);
    c = wmma_bf16(load_a_frag_lds<TP>(sARb, mt * 16, 0, lane),
                  load_b_frag_lds<TP>(sPIT, nt * 16, lane), c);
    const int n  = nt * 16 + (lane & 15);
    const float bb = bO1[n] + bI1[n];
    const int tcol = mt * 16 + ((lane & 16) ? 8 : 0);
    __bf16* p = sH1T + n * TP + tcol;
#pragma unroll
    for (int j = 0; j < 8; j += 2) {
      BF2U q;
      q.h.lo = (__bf16)fmaxf(c[j] + bb, 0.0f);
      q.h.hi = (__bf16)fmaxf(c[j + 1] + bb, 0.0f);
      *(unsigned*)(p + j) = q.u;
    }
  }
  __syncthreads();

  // ---- Step 5: ho1 = A_f h1, hi1 = A_r h1 (32x64) -> A-operand bf16 -------
  {
#pragma unroll
    for (int q = 0; q < 2; ++q) {
      const int job = wave * 2 + q;           // 16 tile jobs
      const int dir = job >> 3;               // uniform per wave
      const int mt  = (job >> 2) & 1;
      const int nt  = job & 3;
      v8f c = {};
      c = wmma_bf16(load_a_frag_lds<TP>(dir ? sARb : sAFb, mt * 16, 0, lane),
                    load_b_frag_lds<TP>(sH1T, nt * 16, lane), c);
      __bf16* dst = dir ? sHI1 : sHO1;
      const int col = nt * 16 + (lane & 15);
      const int rb  = mt * 16 + ((lane & 16) ? 8 : 0);
#pragma unroll
      for (int j = 0; j < 8; ++j)
        dst[(rb + j) * AP + col] = (__bf16)c[j];
    }
  }
  __syncthreads();

  // ---- Step 6: h2 = relu(ho1@WO2^T + hi1@WI2^T + b2) -> sOut (f32) --------
  {
    const int m_base = (wave & 1) * 16;
    const int nt0    = (wave >> 1) * 4;
    const v16bf a00 = load_a_frag_lds<AP>(sHO1, m_base, 0, lane);
    const v16bf a01 = load_a_frag_lds<AP>(sHO1, m_base, 32, lane);
    const v16bf a10 = load_a_frag_lds<AP>(sHI1, m_base, 0, lane);
    const v16bf a11 = load_a_frag_lds<AP>(sHI1, m_base, 32, lane);
#pragma unroll
    for (int t = 0; t < 4; ++t) {
      const int nt = nt0 + t;
      v8f c = {};
      c = wmma_bf16(a00, load_b_frag_pre(WO2f, 2, nt, 0, lane), c);
      c = wmma_bf16(a01, load_b_frag_pre(WO2f, 2, nt, 1, lane), c);
      c = wmma_bf16(a10, load_b_frag_pre(WI2f, 2, nt, 0, lane), c);
      c = wmma_bf16(a11, load_b_frag_pre(WI2f, 2, nt, 1, lane), c);
      const int col  = nt * 16 + (lane & 15);
      const float bb = bO2[col] + bI2[col];
      const int rb   = m_base + ((lane & 16) ? 8 : 0);
#pragma unroll
      for (int j = 0; j < 8; ++j)
        sOut[(rb + j) * DD + col] = fmaxf(c[j] + bb, 0.0f);
    }
  }
  __syncthreads();

  // ---- mean readout over the 32 nodes -------------------------------------
  {
    float sum = 0.f;
#pragma unroll
    for (int r = 0; r < NPG; ++r) sum += sOut[r * DD + tid];
    out_embs[((size_t)side * BG + g) * DD + tid] = sum * (1.0f / NPG);
  }
}

// ---------------------------------------------------------------------------
// Kernel 2: r = relu(rel_tok @ relW^T + relb). One block per 16-row M tile.
// ---------------------------------------------------------------------------
__global__ __launch_bounds__(256) void rel_kernel(
    const float* __restrict__ rel_tok, const __bf16* __restrict__ relWf,
    const float* __restrict__ relb, float* __restrict__ r_out) {
  __shared__ __attribute__((aligned(16))) __bf16 sRB[16 * HP];
  const int tid = threadIdx.x, lane = tid & 31, wave = tid >> 5;
  const int bm = blockIdx.x;

  const float* gsrc = rel_tok + (size_t)bm * 16 * DD;
#pragma unroll
  for (int i = 0; i < 8; ++i) {
    const int idx = i * 512 + tid * 2;
    const int row = idx >> 8;
    const int col = idx & 255;
    const float2 q = *(const float2*)(gsrc + idx);
    BF2U p;
    p.h.lo = (__bf16)q.x; p.h.hi = (__bf16)q.y;
    *(unsigned*)&sRB[row * HP + col] = p.u;
  }
  __syncthreads();

  v8f c0 = {}, c1 = {};
  const int nt0 = wave * 2;
#pragma unroll
  for (int kk = 0; kk < 8; ++kk) {
    const v16bf a = load_a_frag_lds<HP>(sRB, 0, kk * 32, lane);
    c0 = wmma_bf16(a, load_b_frag_pre(relWf, 8, nt0, kk, lane), c0);
    c1 = wmma_bf16(a, load_b_frag_pre(relWf, 8, nt0 + 1, kk, lane), c1);
  }
  const int col0 = nt0 * 16 + (lane & 15);
  const int col1 = col0 + 16;
  const float b0 = relb[col0], b1 = relb[col1];
  const int rb = (lane & 16) ? 8 : 0;
#pragma unroll
  for (int j = 0; j < 8; ++j) {
    const size_t row = (size_t)bm * 16 + rb + j;
    r_out[row * DD + col0] = fmaxf(c0[j] + b0, 0.0f);
    r_out[row * DD + col1] = fmaxf(c1[j] + b1, 0.0f);
  }
}

// ---------------------------------------------------------------------------
// Kernel 3: pos/neg L2 distances. One block per graph triple.
// ---------------------------------------------------------------------------
__global__ __launch_bounds__(256) void dist_kernel(
    const float* __restrict__ ws, const int* __restrict__ chid,
    const int* __restrict__ ctid, float* __restrict__ out) {
  __shared__ float rp[256];
  __shared__ float rn[256];
  const int tid = threadIdx.x;
  const int g = blockIdx.x;
  const float* h = ws;
  const float* t = ws + (size_t)BG * DD;
  const float* r = ws + (size_t)2 * BG * DD;

  const float rv = r[(size_t)g * DD + tid];
  const float pd = h[(size_t)g * DD + tid] + rv - t[(size_t)g * DD + tid];
  const int ch = chid[g], ct = ctid[g];
  const float nd = h[(size_t)ch * DD + tid] + rv - t[(size_t)ct * DD + tid];
  rp[tid] = pd * pd;
  rn[tid] = nd * nd;
  __syncthreads();
#pragma unroll
  for (int s = 128; s > 0; s >>= 1) {
    if (tid < s) { rp[tid] += rp[tid + s]; rn[tid] += rn[tid + s]; }
    __syncthreads();
  }
  if (tid == 0) {
    out[g]      = sqrtf(rp[0]);
    out[BG + g] = sqrtf(rn[0]);
  }
}

// ---------------------------------------------------------------------------
extern "C" void kernel_launch(void* const* d_in, const int* in_sizes, int n_in,
                              void* d_out, int out_size, void* d_ws, size_t ws_size,
                              hipStream_t stream) {
  (void)in_sizes; (void)n_in; (void)out_size; (void)ws_size;
  const float* subj_nodes = (const float*)d_in[0];
  const float* obj_nodes  = (const float*)d_in[1];
  const float* rel_tok    = (const float*)d_in[2];
  const int*   subj_src   = (const int*)d_in[3];
  const int*   subj_dst   = (const int*)d_in[4];
  const int*   obj_src    = (const int*)d_in[5];
  const int*   obj_dst    = (const int*)d_in[6];
  /* d_in[7] node_graph_ids: implied by layout */
  const int*   chid       = (const int*)d_in[8];
  const int*   ctid       = (const int*)d_in[9];
  const float* WO1  = (const float*)d_in[10];
  const float* bO1  = (const float*)d_in[11];
  const float* WI1  = (const float*)d_in[12];
  const float* bI1  = (const float*)d_in[13];
  const float* WO2  = (const float*)d_in[14];
  const float* bO2  = (const float*)d_in[15];
  const float* WI2  = (const float*)d_in[16];
  const float* bI2  = (const float*)d_in[17];
  const float* relW = (const float*)d_in[18];
  const float* relb = (const float*)d_in[19];

  float* ws = (float*)d_ws;
  float* embs = ws;                              // h then t (2*BG*256)
  float* rE   = ws + (size_t)2 * BG * DD;        // r (BG*256)
  __bf16* wf   = (__bf16*)(ws + (size_t)3 * BG * DD);
  __bf16* WO1f = wf;                             // 64*256
  __bf16* WI1f = WO1f + 64 * 256;
  __bf16* WO2f = WI1f + 64 * 256;                // 256*64
  __bf16* WI2f = WO2f + 256 * 64;
  __bf16* relWf = WI2f + 256 * 64;               // 256*256

  prep_weight_kernel<<<4, 256, 0, stream>>>(WO1, WO1f, 4, 8, 256);
  prep_weight_kernel<<<4, 256, 0, stream>>>(WI1, WI1f, 4, 8, 256);
  prep_weight_kernel<<<4, 256, 0, stream>>>(WO2, WO2f, 16, 2, 64);
  prep_weight_kernel<<<4, 256, 0, stream>>>(WI2, WI2f, 16, 2, 64);
  prep_weight_kernel<<<16, 256, 0, stream>>>(relW, relWf, 16, 8, 256);

  graph_emb_kernel<<<2 * BG, 256, 0, stream>>>(
      subj_nodes, obj_nodes, subj_src, subj_dst, obj_src, obj_dst,
      WO1f, bO1, WI1f, bI1, WO2f, bO2, WI2f, bI2, embs);

  rel_kernel<<<BG / 16, 256, 0, stream>>>(rel_tok, relWf, relb, rE);

  dist_kernel<<<BG, 256, 0, stream>>>(ws, chid, ctid, (float*)d_out);
}
